// TabNSA_40321152975419
// MI455X (gfx1250) — matrix operand order repaired and checked by reading, not compiled, not run on hardware
//
#include <hip/hip_runtime.h>
#include <hip/hip_bf16.h>

// ---------------------------------------------------------------------------
// Problem constants (from reference)
// ---------------------------------------------------------------------------
#define BB 16384
#define DD 1024
#define HH 512
#define FFD 1024
#define LL 4
#define KTOP 64

typedef __attribute__((ext_vector_type(16))) __bf16 v16bf;
typedef __attribute__((ext_vector_type(8)))  __bf16 v8bf;
typedef __attribute__((ext_vector_type(8)))  float  v8f;

static __device__ __forceinline__ __bf16 f2bf(float f) {
    unsigned int u = __builtin_bit_cast(unsigned int, f);
    unsigned int r = (u + 0x7FFFu + ((u >> 16) & 1u)) >> 16;
    unsigned short us = (unsigned short)r;
    return __builtin_bit_cast(__bf16, us);
}

// ---------------------------------------------------------------------------
// fp32 -> bf16 cast
// ---------------------------------------------------------------------------
__global__ void cast_f32_bf16(const float* __restrict__ in,
                              __bf16* __restrict__ out, long n) {
    long i = (long)blockIdx.x * blockDim.x + threadIdx.x;
    if (i < n) out[i] = f2bf(in[i]);
}

// ---------------------------------------------------------------------------
// WMMA bf16 GEMM:  C[M,N] = act( (A[M,K] @ W[N,K]^T + bias) * scale + resid )
// Each wave computes a 64x64 macro-tile = 4x4 grid of 16x16 WMMA tiles with
// V_WMMA_F32_16X16X32_BF16; 4 A-fragments and 4 B-fragments are loaded per
// 32-wide K-step and reused 4x each (16 WMMAs per 8 fragment loads).
// Prefetches are unconditional (speculative prefetch is dropped silently on
// bad translation per ISA 7.3) at +64 elems = one 128B line ahead.
// A, W row-major with K contiguous (weights stored (N,K) => W rows are B cols).
// ---------------------------------------------------------------------------
#define MT 4
#define NT 4
__global__ __launch_bounds__(256) void wmma_gemm_bf16(
    const __bf16* __restrict__ A, int lda,
    const __bf16* __restrict__ W, int ldw,
    const float* __restrict__ bias,
    const float* __restrict__ resid,
    float scale, int relu,
    float* __restrict__ Cf, __bf16* __restrict__ Cb,
    int M, int N, int K)
{
    int lane   = threadIdx.x & 31;
    long wid   = (long)blockIdx.x * 8 + (threadIdx.x >> 5);
    int tilesN = N >> 6;                       // 64-wide macro tiles in N
    long total = (long)(M >> 6) * tilesN;
    if (wid >= total) return;
    int tm = (int)(wid / tilesN) << 6;
    int tn = (int)(wid % tilesN) << 6;

    int l15   = lane & 15;
    int khalf = lane >> 4;                     // 0 or 1

    const __bf16* Ar[MT];
    const __bf16* Br[NT];
#pragma unroll
    for (int i = 0; i < MT; ++i) Ar[i] = A + (long)(tm + i * 16 + l15) * lda;
#pragma unroll
    for (int j = 0; j < NT; ++j) Br[j] = W + (long)(tn + j * 16 + l15) * ldw;

    v8f acc[MT][NT] = {};

#pragma unroll 1
    for (int k0 = 0; k0 < K; k0 += 32) {
        v16bf af[MT], bf[NT];
        // A fragment 16x32: lanes 0-15 hold K=k0..k0+15, lanes 16-31 K=+16..+31
#pragma unroll
        for (int i = 0; i < MT; ++i) {
            af[i] = *(const v16bf*)(Ar[i] + k0 + 16 * khalf);
            __builtin_prefetch(Ar[i] + k0 + 64, 0, 3);   // global_prefetch_b8
        }
        // B fragment 32x16: lanes 0-15 hold K=k0..k0+15, lanes 16-31 K=+16..+31
#pragma unroll
        for (int j = 0; j < NT; ++j) {
            bf[j] = *(const v16bf*)(Br[j] + k0 + 16 * khalf);
            __builtin_prefetch(Br[j] + k0 + 64, 0, 3);
        }
#pragma unroll
        for (int i = 0; i < MT; ++i)
#pragma unroll
            for (int j = 0; j < NT; ++j)
                acc[i][j] = __builtin_amdgcn_wmma_f32_16x16x32_bf16(
                    false, af[i], false, bf[j], (short)0, acc[i][j], false, false);
    }

    // C/D layout: lane in [0,15]: M=r, N=lane; lane in [16,31]: M=8+r, N=lane-16
#pragma unroll
    for (int j = 0; j < NT; ++j) {
        int n = tn + j * 16 + l15;
        float bn = bias ? bias[n] : 0.0f;
#pragma unroll
        for (int i = 0; i < MT; ++i) {
#pragma unroll
            for (int r = 0; r < 8; ++r) {
                int m = tm + i * 16 + r + (khalf << 3);
                float v = (acc[i][j][r] + bn) * scale;
                long idx = (long)m * N + n;
                if (resid) v += resid[idx];
                if (relu)  v = fmaxf(v, 0.0f);
                if (Cf) Cf[idx] = v;
                if (Cb) Cb[idx] = f2bf(v);
            }
        }
    }
}

// ---------------------------------------------------------------------------
// Per-row softmax over 1024 logits, top-64 threshold by bisection,
// emit xm[b,d] = x[b,d]*prob[b,d] for selected entries (bf16), else 0.
// ---------------------------------------------------------------------------
__global__ __launch_bounds__(256) void softmax_topk_mask(
    const float* __restrict__ logits, const float* __restrict__ x,
    __bf16* __restrict__ xm)
{
    __shared__ float p[DD];
    __shared__ float red[256];
    __shared__ float sh_lo, sh_hi;
    int row = blockIdx.x;
    int t   = threadIdx.x;
    const float* lrow = logits + (long)row * DD;

    float lmax = -1e30f;
#pragma unroll
    for (int j = 0; j < 4; ++j) {
        float v = lrow[t + 256 * j];
        p[t + 256 * j] = v;
        lmax = fmaxf(lmax, v);
    }
    red[t] = lmax; __syncthreads();
    for (int s = 128; s > 0; s >>= 1) { if (t < s) red[t] = fmaxf(red[t], red[t + s]); __syncthreads(); }
    float rmax = red[0]; __syncthreads();

    float lsum = 0.0f;
#pragma unroll
    for (int j = 0; j < 4; ++j) {
        float e = __expf(p[t + 256 * j] - rmax);
        p[t + 256 * j] = e;
        lsum += e;
    }
    red[t] = lsum; __syncthreads();
    for (int s = 128; s > 0; s >>= 1) { if (t < s) red[t] += red[t + s]; __syncthreads(); }
    float inv = 1.0f / red[0]; __syncthreads();

#pragma unroll
    for (int j = 0; j < 4; ++j) p[t + 256 * j] *= inv;
    if (t == 0) { sh_lo = 0.0f; sh_hi = inv; }   // max prob == inv
    __syncthreads();

    // bisect for the 64th-largest prob: invariant count(>lo) >= 64 > count(>hi)
    for (int it = 0; it < 32; ++it) {
        float mid = 0.5f * (sh_lo + sh_hi);
        int c = 0;
#pragma unroll
        for (int j = 0; j < 4; ++j) c += (p[t + 256 * j] > mid) ? 1 : 0;
        red[t] = (float)c; __syncthreads();
        for (int s = 128; s > 0; s >>= 1) { if (t < s) red[t] += red[t + s]; __syncthreads(); }
        if (t == 0) { if (red[0] >= (float)KTOP) sh_lo = mid; else sh_hi = mid; }
        __syncthreads();
    }
    float thr = sh_lo;
    const float* xrow = x + (long)row * DD;
    __bf16* orow = xm + (long)row * DD;
#pragma unroll
    for (int j = 0; j < 4; ++j) {
        int idx = t + 256 * j;
        float pr = p[idx];
        orow[idx] = f2bf((pr > thr) ? xrow[idx] * pr : 0.0f);
    }
}

// ---------------------------------------------------------------------------
// LayerNorm over 512 cols; writes fp32 (residual source) + bf16 (GEMM input)
// ---------------------------------------------------------------------------
__global__ __launch_bounds__(256) void layernorm_512(
    const float* __restrict__ in, const float* __restrict__ g,
    const float* __restrict__ b, float* __restrict__ outf,
    __bf16* __restrict__ outb)
{
    __shared__ float red[256];
    int row = blockIdx.x, t = threadIdx.x;
    const float* irow = in + (long)row * HH;
    float v0 = irow[t], v1 = irow[t + 256];
    red[t] = v0 + v1; __syncthreads();
    for (int s = 128; s > 0; s >>= 1) { if (t < s) red[t] += red[t + s]; __syncthreads(); }
    float mean = red[0] * (1.0f / (float)HH); __syncthreads();
    float d0 = v0 - mean, d1 = v1 - mean;
    red[t] = d0 * d0 + d1 * d1; __syncthreads();
    for (int s = 128; s > 0; s >>= 1) { if (t < s) red[t] += red[t + s]; __syncthreads(); }
    float rstd = rsqrtf(red[0] * (1.0f / (float)HH) + 1e-5f);
    float o0 = d0 * rstd * g[t] + b[t];
    float o1 = d1 * rstd * g[t + 256] + b[t + 256];
    long base = (long)row * HH;
    outf[base + t] = o0;        outf[base + t + 256] = o1;
    outb[base + t] = f2bf(o0);  outb[base + t + 256] = f2bf(o1);
}

// ---------------------------------------------------------------------------
// out[b] = h[b,:] . Wout + bout   (wave-per-row GEMV, wave32 shuffle reduce)
// ---------------------------------------------------------------------------
__global__ __launch_bounds__(256) void out_gemv(
    const float* __restrict__ h, const float* __restrict__ wout,
    const float* __restrict__ bout, float* __restrict__ out)
{
    int gid  = blockIdx.x * 256 + threadIdx.x;
    int row  = gid >> 5;
    int lane = gid & 31;
    if (row >= BB) return;
    const float* hr = h + (long)row * HH;
    float s = 0.0f;
    for (int j = lane; j < HH; j += 32) s += hr[j] * wout[j];
#pragma unroll
    for (int off = 16; off; off >>= 1) s += __shfl_xor(s, off, 32);
    if (lane == 0) out[row] = s + bout[0];
}

// ---------------------------------------------------------------------------
// Orchestration
// ---------------------------------------------------------------------------
extern "C" void kernel_launch(void* const* d_in, const int* in_sizes, int n_in,
                              void* d_out, int out_size, void* d_ws, size_t ws_size,
                              hipStream_t stream) {
    const float* x     = (const float*)d_in[0];
    const float* Ws    = (const float*)d_in[1];
    const float* bs    = (const float*)d_in[2];
    const float* Wp    = (const float*)d_in[3];
    const float* bp    = (const float*)d_in[4];
    const float* Wqkv  = (const float*)d_in[5];
    const float* bqkv  = (const float*)d_in[6];
    const float* Wo    = (const float*)d_in[7];
    const float* bo    = (const float*)d_in[8];
    const float* W1    = (const float*)d_in[9];
    const float* b1    = (const float*)d_in[10];
    const float* W2    = (const float*)d_in[11];
    const float* b2    = (const float*)d_in[12];
    const float* ln1_g = (const float*)d_in[13];
    const float* ln1_b = (const float*)d_in[14];
    const float* ln2_g = (const float*)d_in[15];
    const float* ln2_b = (const float*)d_in[16];
    const float* Wout  = (const float*)d_in[17];
    const float* bout  = (const float*)d_in[18];
    float* out = (float*)d_out;

    size_t off = 0;
    auto alloc = [&](size_t bytes) -> void* {
        void* p = (char*)d_ws + off;
        off += (bytes + 255) & ~(size_t)255;
        return p;
    };
    __bf16* xb     = (__bf16*)alloc((size_t)BB * DD * 2);
    __bf16* wsb    = (__bf16*)alloc((size_t)DD * DD * 2);
    __bf16* wpb    = (__bf16*)alloc((size_t)HH * DD * 2);
    __bf16* wvb    = (__bf16*)alloc((size_t)LL * HH * HH * 2);
    __bf16* wob    = (__bf16*)alloc((size_t)LL * HH * HH * 2);
    __bf16* w1b    = (__bf16*)alloc((size_t)LL * FFD * HH * 2);
    __bf16* w2b    = (__bf16*)alloc((size_t)LL * HH * FFD * 2);
    float*  logits = (float* )alloc((size_t)BB * DD * 4);
    __bf16* xmb    = (__bf16*)alloc((size_t)BB * DD * 2);
    float*  hf     = (float* )alloc((size_t)BB * HH * 4);
    __bf16* hb     = (__bf16*)alloc((size_t)BB * HH * 2);
    __bf16* t1b    = (__bf16*)alloc((size_t)BB * FFD * 2);
    float*  pre    = (float* )alloc((size_t)BB * HH * 4);

    auto cast = [&](const float* in, __bf16* o, long n) {
        cast_f32_bf16<<<(unsigned)((n + 255) / 256), 256, 0, stream>>>(in, o, n);
    };
    cast(x,   xb,  (long)BB * DD);
    cast(Ws,  wsb, (long)DD * DD);
    cast(Wp,  wpb, (long)HH * DD);
    for (int i = 0; i < LL; ++i)   // V slice of Wqkv: rows [2H, 3H)
        cast(Wqkv + (size_t)i * 3 * HH * HH + (size_t)2 * HH * HH,
             wvb + (size_t)i * HH * HH, (long)HH * HH);
    cast(Wo,  wob, (long)LL * HH * HH);
    cast(W1,  w1b, (long)LL * FFD * HH);
    cast(W2,  w2b, (long)LL * HH * FFD);

    auto gemm = [&](const __bf16* A, int lda, const __bf16* W, int ldw,
                    const float* bias, const float* resid, float scale, int relu,
                    float* Cf, __bf16* Cb, int M, int N, int K) {
        long tiles = (long)(M / 64) * (N / 64);      // 64x64 macro tiles
        wmma_gemm_bf16<<<(unsigned)((tiles + 7) / 8), 256, 0, stream>>>(
            A, lda, W, ldw, bias, resid, scale, relu, Cf, Cb, M, N, K);
    };

    // 1) logits = (x @ Ws^T + bs) / sqrt(D)
    gemm(xb, DD, wsb, DD, bs, nullptr, 1.0f / 32.0f, 0, logits, nullptr, BB, DD, DD);
    // 2) softmax + top-64 mask -> xm = x * mask (bf16)
    softmax_topk_mask<<<BB, 256, 0, stream>>>(logits, x, xmb);
    // 3) h = xm @ Wp^T + bp
    gemm(xmb, DD, wpb, DD, bp, nullptr, 1.0f, 0, hf, hb, BB, HH, DD);

    for (int i = 0; i < LL; ++i) {
        const __bf16* wv = wvb + (size_t)i * HH * HH;
        const __bf16* wo = wob + (size_t)i * HH * HH;
        const __bf16* w1 = w1b + (size_t)i * FFD * HH;
        const __bf16* w2 = w2b + (size_t)i * HH * FFD;
        const float* bv  = bqkv + (size_t)i * 3 * HH + 2 * HH;
        // v = h @ Wv^T + bv  (bf16 only)
        gemm(hb, HH, wv, HH, bv, nullptr, 1.0f, 0, nullptr, t1b, BB, HH, HH);
        // pre = v @ Wo^T + bo + h
        gemm(t1b, HH, wo, HH, bo + (size_t)i * HH, hf, 1.0f, 0, pre, nullptr, BB, HH, HH);
        // h = LN1(pre)
        layernorm_512<<<BB, 256, 0, stream>>>(pre, ln1_g + (size_t)i * HH,
                                              ln1_b + (size_t)i * HH, hf, hb);
        // f = relu(h @ W1^T + b1)  (bf16 only)
        gemm(hb, HH, w1, HH, b1 + (size_t)i * FFD, nullptr, 1.0f, 1, nullptr, t1b, BB, FFD, HH);
        // pre = f @ W2^T + b2 + h
        gemm(t1b, FFD, w2, FFD, b2 + (size_t)i * HH, hf, 1.0f, 0, pre, nullptr, BB, HH, FFD);
        // h = LN2(pre)
        layernorm_512<<<BB, 256, 0, stream>>>(pre, ln2_g + (size_t)i * HH,
                                              ln2_b + (size_t)i * HH, hf, hb);
    }
    // out = h @ Wout^T + bout
    out_gemv<<<(BB * 32 + 255) / 256, 256, 0, stream>>>(hf, Wout, bout, out);
}
